// ResidualBlock_3667902071215
// MI455X (gfx1250) — matrix operand order repaired and checked by reading, not compiled, not run on hardware
//
#include <hip/hip_runtime.h>
#include <hip/hip_bf16.h>
#include <cmath>

// ---------------------------------------------------------------------------
// Mamba block for MI455X (gfx1250, wave32, WMMA).
// GEMMs: bf16 WMMA (v_wmma_f32_16x16x32_bf16), f32 accumulate.
// Block 64x256, K-step 64 (16 WMMAs per barrier period), 8 wave32 per block.
// Staging loads are batched into distinct registers for full MLP.
// Workspace requirement: ~225 MB (bump allocator in kernel_launch).
// ---------------------------------------------------------------------------

typedef __attribute__((ext_vector_type(16))) __bf16 v16bf;
typedef __attribute__((ext_vector_type(8)))  __bf16 v8bf;
typedef __attribute__((ext_vector_type(2)))  __bf16 v2bf;
typedef __attribute__((ext_vector_type(8)))  float  v8f;

#define D_MODEL 1024
#define D_INNER 2048
#define D_STATE 16
#define DT_RANK 64
#define D_CONV  4
#define EPS     1e-5f
#define B_SZ    2
#define SEQ_L   2048
#define ROWS    (B_SZ * SEQ_L)          // 4096

// ------------------------------- RMSNorm -----------------------------------
__global__ __launch_bounds__(256)
void rmsnorm_bf16_kernel(const float* __restrict__ x,
                         const float* __restrict__ w,
                         __bf16* __restrict__ out) {
  __shared__ float red[256];
  const int row = blockIdx.x;
  const float* xr = x + (size_t)row * D_MODEL;
  float ss = 0.f;
  for (int c = threadIdx.x; c < D_MODEL; c += 256) { float v = xr[c]; ss += v * v; }
  red[threadIdx.x] = ss;
  __syncthreads();
  for (int s = 128; s > 0; s >>= 1) {
    if (threadIdx.x < s) red[threadIdx.x] += red[threadIdx.x + s];
    __syncthreads();
  }
  const float scale = rsqrtf(red[0] * (1.0f / D_MODEL) + EPS);
  __bf16* orow = out + (size_t)row * D_MODEL;
  for (int c = threadIdx.x; c < D_MODEL; c += 256)
    orow[c] = (__bf16)(xr[c] * scale * w[c]);
}

// --------------------------- f32 -> bf16 convert ---------------------------
__global__ __launch_bounds__(256)
void f32_to_bf16_2d(const float* __restrict__ src, int sld,
                    __bf16* __restrict__ dst, int dld,
                    int rows, int cols) {
  const int idx = blockIdx.x * 256 + threadIdx.x;
  if (idx >= rows * cols) return;
  const int r = idx / cols, c = idx % cols;
  dst[(size_t)r * dld + c] = (__bf16)src[(size_t)r * sld + c];
}

// ------------------------------ WMMA GEMM ----------------------------------
// C(M,N) = A(M,K) x B(K,N), row-major, A/B bf16, C f32.
// 256 threads = 8 wave32 (2x4 wave grid); block tile 64x256; K-step 64
// (two 32-K substeps -> 16 WMMAs per staged tile).
// FULL_TILE: compile out all M/N guards (tiles exact multiples).
// epilogue: 0 = store, 1 = softplus(acc + bias[n]), 2 = acc + resid[m][n].
#define BM 64
#define BN 256
#define BK 64
#define SAS (BK + 8)                    // 72 halfs = 144 B row stride (16B-aligned)

template <bool FULL_TILE>
__global__ __launch_bounds__(256)
void wmma_gemm_bf16(const __bf16* __restrict__ A, int lda,
                    const __bf16* __restrict__ B, int ldb,
                    float* __restrict__ C, int ldc,
                    int M, int N, int K,
                    int epilogue,
                    const float* __restrict__ bias,
                    const float* __restrict__ resid) {
  __shared__ __bf16 sA[BM][SAS] __attribute__((aligned(16)));
  __shared__ __bf16 sB[BN][SAS] __attribute__((aligned(16)));   // transposed [n][k]

  const int tid  = threadIdx.x;
  const int lane = tid & 31;
  const int wid  = tid >> 5;
  const int wm   = wid >> 2;            // 0..1
  const int wn   = wid & 3;             // 0..3
  const int m0   = blockIdx.y * BM;
  const int n0   = blockIdx.x * BN;

  v8f acc[2][4];
#pragma unroll
  for (int i = 0; i < 2; ++i)
#pragma unroll
    for (int j = 0; j < 4; ++j)
#pragma unroll
      for (int e = 0; e < 8; ++e) acc[i][j][e] = 0.f;

  // A staging: 64 rows x 64 halfs = 512 16B-chunks -> 2 per thread.
  const int aRow   = tid >> 2;          // 0..63
  const int aChunk = (tid & 3) * 8;     // 0,8,16,24 (+32 for second chunk)
  // B staging: 32 k-pairs x 32 n-chunks = 1024 items -> 4 per thread.
  const int nck = (tid & 31) * 8;       // 0..248
  const int kp0 = tid >> 5;             // 0..7
  const bool bValid = FULL_TILE || (n0 + nck) < N;

  for (int k0 = 0; k0 < K; k0 += BK) {
    // ---- batch all 10 staging loads into distinct registers (full MLP) ----
    v8bf aR[2];
#pragma unroll
    for (int h = 0; h < 2; ++h)
      aR[h] = *(const v8bf*)(A + (size_t)(m0 + aRow) * lda + k0 + aChunk + h * 32);

    v8bf bR0[4], bR1[4];
#pragma unroll
    for (int p = 0; p < 4; ++p) {
#pragma unroll
      for (int e = 0; e < 8; ++e) { bR0[p][e] = (__bf16)0.f; bR1[p][e] = (__bf16)0.f; }
      const int kk = (kp0 + p * 8) * 2;                 // even k: 0..62
      if (bValid) {
        bR0[p] = *(const v8bf*)(B + (size_t)(k0 + kk)     * ldb + n0 + nck);
        bR1[p] = *(const v8bf*)(B + (size_t)(k0 + kk + 1) * ldb + n0 + nck);
      }
    }

    // ---- drain registers into LDS ----
#pragma unroll
    for (int h = 0; h < 2; ++h)
      *(v8bf*)&sA[aRow][aChunk + h * 32] = aR[h];
#pragma unroll
    for (int p = 0; p < 4; ++p) {
      const int kk = (kp0 + p * 8) * 2;
#pragma unroll
      for (int e = 0; e < 8; ++e) {
        v2bf pk; pk[0] = bR0[p][e]; pk[1] = bR1[p][e];
        *(v2bf*)&sB[nck + e][kk] = pk;                  // aligned ds_store_b32
      }
    }

    // ---- prefetch next K tile (global_prefetch_b8) ----
    if (k0 + BK < K) {
      __builtin_prefetch(A + (size_t)(m0 + aRow) * lda + k0 + BK + aChunk, 0, 1);
      if (bValid)
        __builtin_prefetch(B + (size_t)(k0 + BK + kp0 * 2) * ldb + n0 + nck, 0, 1);
    }
    __syncthreads();

    // ---- two 32-K substeps; fragments per CDNA5 16-bit VGPR layouts ----
    // A 16x32: lane m = lane%16; elems [0..7] = K (lane/16)*8..+7,
    //          elems [8..15] = K 16+(lane/16)*8..+7     -> two ds_load_b128.
    // B 32x16 (from transposed LDS): lane n = lane%16;
    //          elems [0..7] = K (lane/16)*8..+7, [8..15] = K 16+(lane/16)*8..+7 -> two ds_load_b128.
    const int fm = lane & 15;
    const int kh = (lane >> 4) * 8;     // 0 or 8
    const int ks = (lane >> 4) * 16;    // 0 or 16

#pragma unroll
    for (int ks0 = 0; ks0 < BK; ks0 += 32) {
      v16bf afrag[2];
#pragma unroll
      for (int im = 0; im < 2; ++im) {
        const int r = wm * 32 + im * 16 + fm;
        v8bf lo = *(const v8bf*)&sA[r][ks0 + kh];
        v8bf hi = *(const v8bf*)&sA[r][ks0 + 16 + kh];
#pragma unroll
        for (int e = 0; e < 8; ++e) { afrag[im][e] = lo[e]; afrag[im][8 + e] = hi[e]; }
      }
      v16bf bfrag[4];
#pragma unroll
      for (int jn = 0; jn < 4; ++jn) {
        const int r = wn * 64 + jn * 16 + fm;
        v8bf c0 = *(const v8bf*)&sB[r][ks0 + kh];
        v8bf c1 = *(const v8bf*)&sB[r][ks0 + 16 + kh];
#pragma unroll
        for (int e = 0; e < 8; ++e) { bfrag[jn][e] = c0[e]; bfrag[jn][8 + e] = c1[e]; }
      }
#pragma unroll
      for (int im = 0; im < 2; ++im)
#pragma unroll
        for (int jn = 0; jn < 4; ++jn)
          acc[im][jn] = __builtin_amdgcn_wmma_f32_16x16x32_bf16(
              false, afrag[im], false, bfrag[jn],
              (short)0, acc[im][jn], false, false);
    }
    __syncthreads();
  }

  // ---- epilogue: C/D layout — lane n = lane%16, VGPR r -> m = (lane/16)*8 + r
#pragma unroll
  for (int im = 0; im < 2; ++im) {
#pragma unroll
    for (int jn = 0; jn < 4; ++jn) {
      const int n = n0 + wn * 64 + jn * 16 + (lane & 15);
      if (!FULL_TILE && n >= N) continue;
#pragma unroll
      for (int r = 0; r < 8; ++r) {
        const int m = m0 + wm * 32 + im * 16 + (lane >> 4) * 8 + r;
        float v = acc[im][jn][r];
        if (epilogue == 1) {
          v += bias[n];
          v = (v > 20.f) ? v : log1pf(__expf(v));       // softplus
        } else if (epilogue == 2) {
          v += resid[(size_t)m * ldc + n];
        }
        C[(size_t)m * ldc + n] = v;
      }
    }
  }
}

// ----------------- causal depthwise conv1d (k=4) + SiLU --------------------
__global__ __launch_bounds__(256)
void conv_silu_kernel(const float* __restrict__ xi_res,   // (ROWS, 2*D_INNER), xi = cols [0,2048)
                      const float* __restrict__ cw,       // (D_INNER, 1, 4)
                      const float* __restrict__ cb,
                      float* __restrict__ xi_f32,
                      __bf16* __restrict__ xi_bf) {
  const int idx = blockIdx.x * 256 + threadIdx.x;
  if (idx >= ROWS * D_INNER) return;
  const int d   = idx % D_INNER;
  const int row = idx / D_INNER;        // b*L + l
  const int l   = row % SEQ_L;
  float acc = cb[d];
#pragma unroll
  for (int j = 0; j < D_CONV; ++j) {
    const int ll = l - (D_CONV - 1) + j;
    if (ll >= 0)
      acc += cw[d * D_CONV + j] *
             xi_res[(size_t)(row - (D_CONV - 1) + j) * (2 * D_INNER) + d];
  }
  const float s = acc / (1.f + __expf(-acc));             // SiLU
  xi_f32[(size_t)row * D_INNER + d] = s;
  xi_bf [(size_t)row * D_INNER + d] = (__bf16)s;
}

// --------------------------- selective scan --------------------------------
// One thread per (b, d): 16-state recurrence in registers, sequential over L.
// B/C (shared across d) staged through LDS in 32-step chunks.
#define TCH 32
__global__ __launch_bounds__(256)
void scan_kernel(const float* __restrict__ x_dbl,   // (ROWS, 96): [.,64:80)=B, [.,80:96)=C
                 const float* __restrict__ delta,   // (ROWS, D_INNER), post-softplus
                 const float* __restrict__ xi,      // (ROWS, D_INNER)
                 const float* __restrict__ A_log,   // (D_INNER, 16)
                 const float* __restrict__ Dp,      // (D_INNER)
                 float* __restrict__ y) {           // (ROWS, D_INNER)
  __shared__ float sB[TCH][D_STATE];
  __shared__ float sC[TCH][D_STATE];
  const int b = blockIdx.x / (D_INNER / 256);
  const int d = (blockIdx.x % (D_INNER / 256)) * 256 + threadIdx.x;

  float Arow[D_STATE], s[D_STATE];
#pragma unroll
  for (int n = 0; n < D_STATE; ++n) {
    Arow[n] = -__expf(A_log[d * D_STATE + n]);
    s[n] = 0.f;
  }
  const float dskip = Dp[d];

  for (int tc = 0; tc < SEQ_L; tc += TCH) {
    for (int i = threadIdx.x; i < TCH * D_STATE; i += 256) {
      const int t = i / D_STATE, n = i % D_STATE;
      const float* xr = x_dbl + (size_t)(b * SEQ_L + tc + t) * (DT_RANK + 2 * D_STATE);
      sB[t][n] = xr[DT_RANK + n];
      sC[t][n] = xr[DT_RANK + D_STATE + n];
    }
    __syncthreads();
    for (int t = 0; t < TCH; ++t) {
      const size_t off = (size_t)(b * SEQ_L + tc + t) * D_INNER + d;
      const float dt = delta[off];
      const float u  = xi[off];
      const float du = dt * u;
      float yv = 0.f;
#pragma unroll
      for (int n = 0; n < D_STATE; ++n) {
        s[n] = __expf(dt * Arow[n]) * s[n] + du * sB[t][n];   // v_exp_f32 TRANS co-exec
        yv  += s[n] * sC[t][n];
      }
      y[off] = yv + u * dskip;
    }
    __syncthreads();
  }
}

// ------------------------------- gating ------------------------------------
__global__ __launch_bounds__(256)
void gate_kernel(const float* __restrict__ ysc,
                 const float* __restrict__ xi_res,   // res = cols [2048,4096)
                 __bf16* __restrict__ ybf) {
  const int idx = blockIdx.x * 256 + threadIdx.x;
  if (idx >= ROWS * D_INNER) return;
  const int row = idx / D_INNER, d = idx % D_INNER;
  const float r = xi_res[(size_t)row * (2 * D_INNER) + D_INNER + d];
  const float g = r / (1.f + __expf(-r));
  ybf[idx] = (__bf16)(ysc[idx] * g);
}

// ---------------------------------------------------------------------------
extern "C" void kernel_launch(void* const* d_in, const int* in_sizes, int n_in,
                              void* d_out, int out_size, void* d_ws, size_t ws_size,
                              hipStream_t stream) {
  const float* x       = (const float*)d_in[0];
  const float* in_w    = (const float*)d_in[1];
  const float* conv_w  = (const float*)d_in[2];
  const float* conv_b  = (const float*)d_in[3];
  const float* xproj_w = (const float*)d_in[4];
  const float* dt_w    = (const float*)d_in[5];
  const float* dt_b    = (const float*)d_in[6];
  const float* A_log   = (const float*)d_in[7];
  const float* Dp      = (const float*)d_in[8];
  const float* out_w   = (const float*)d_in[9];
  const float* norm_w  = (const float*)d_in[10];
  float* out = (float*)d_out;

  char* ws = (char*)d_ws;
  size_t off = 0;
  auto alloc = [&](size_t bytes) -> char* {
    char* p = ws + off;
    off = (off + bytes + 255) & ~(size_t)255;
    return p;
  };
  const int NXP = DT_RANK + 2 * D_STATE;                       // 96
  __bf16* xn_bf    = (__bf16*)alloc((size_t)ROWS * D_MODEL * 2);
  __bf16* w_in_bf  = (__bf16*)alloc((size_t)D_MODEL * 2 * D_INNER * 2);
  float*  xi_res   = (float*) alloc((size_t)ROWS * 2 * D_INNER * 4);
  float*  xi_f32   = (float*) alloc((size_t)ROWS * D_INNER * 4);
  __bf16* xi_bf    = (__bf16*)alloc((size_t)ROWS * D_INNER * 2);
  __bf16* w_xp_bf  = (__bf16*)alloc((size_t)D_INNER * NXP * 2);
  float*  x_dbl    = (float*) alloc((size_t)ROWS * NXP * 4);
  __bf16* dr_bf    = (__bf16*)alloc((size_t)ROWS * DT_RANK * 2);
  __bf16* w_dt_bf  = (__bf16*)alloc((size_t)DT_RANK * D_INNER * 2);
  float*  delta    = (float*) alloc((size_t)ROWS * D_INNER * 4);
  float*  y_scan   = (float*) alloc((size_t)ROWS * D_INNER * 4);
  __bf16* y_bf     = (__bf16*)alloc((size_t)ROWS * D_INNER * 2);
  __bf16* w_out_bf = (__bf16*)alloc((size_t)D_INNER * D_MODEL * 2);
  (void)ws_size; (void)in_sizes; (void)n_in; (void)out_size;

  const int elw = ROWS * D_INNER;
  const int g_elw = (elw + 255) / 256;

  // 1) RMSNorm -> bf16 activations
  rmsnorm_bf16_kernel<<<ROWS, 256, 0, stream>>>(x, norm_w, xn_bf);

  // 2) weight converts to bf16
  f32_to_bf16_2d<<<(D_MODEL * 2 * D_INNER + 255) / 256, 256, 0, stream>>>(
      in_w, 2 * D_INNER, w_in_bf, 2 * D_INNER, D_MODEL, 2 * D_INNER);
  f32_to_bf16_2d<<<(D_INNER * NXP + 255) / 256, 256, 0, stream>>>(
      xproj_w, NXP, w_xp_bf, NXP, D_INNER, NXP);
  f32_to_bf16_2d<<<(DT_RANK * D_INNER + 255) / 256, 256, 0, stream>>>(
      dt_w, D_INNER, w_dt_bf, D_INNER, DT_RANK, D_INNER);
  f32_to_bf16_2d<<<(D_INNER * D_MODEL + 255) / 256, 256, 0, stream>>>(
      out_w, D_MODEL, w_out_bf, D_MODEL, D_INNER, D_MODEL);

  // 3) in_proj: (4096x1024) x (1024x4096) -> xi_res f32   [full tiles]
  {
    dim3 grid((2 * D_INNER) / BN, ROWS / BM);
    wmma_gemm_bf16<true><<<grid, 256, 0, stream>>>(
        xn_bf, D_MODEL, w_in_bf, 2 * D_INNER, xi_res, 2 * D_INNER,
        ROWS, 2 * D_INNER, D_MODEL, 0, nullptr, nullptr);
  }

  // 4) causal depthwise conv + SiLU -> xi (f32 + bf16)
  conv_silu_kernel<<<g_elw, 256, 0, stream>>>(xi_res, conv_w, conv_b, xi_f32, xi_bf);

  // 5) x_proj: (4096x2048) x (2048x96) -> x_dbl f32        [N=96: guarded]
  {
    dim3 grid((NXP + BN - 1) / BN, ROWS / BM);
    wmma_gemm_bf16<false><<<grid, 256, 0, stream>>>(
        xi_bf, D_INNER, w_xp_bf, NXP, x_dbl, NXP,
        ROWS, NXP, D_INNER, 0, nullptr, nullptr);
  }

  // 6) delta_r (cols 0..63 of x_dbl) -> bf16
  f32_to_bf16_2d<<<(ROWS * DT_RANK + 255) / 256, 256, 0, stream>>>(
      x_dbl, NXP, dr_bf, DT_RANK, ROWS, DT_RANK);

  // 7) dt_proj + bias + softplus: (4096x64) x (64x2048) -> delta [full tiles]
  {
    dim3 grid(D_INNER / BN, ROWS / BM);
    wmma_gemm_bf16<true><<<grid, 256, 0, stream>>>(
        dr_bf, DT_RANK, w_dt_bf, D_INNER, delta, D_INNER,
        ROWS, D_INNER, DT_RANK, 1, dt_b, nullptr);
  }

  // 8) selective scan over L
  scan_kernel<<<B_SZ * (D_INNER / 256), 256, 0, stream>>>(
      x_dbl, delta, xi_f32, A_log, Dp, y_scan);

  // 9) gate with silu(res) -> bf16
  gate_kernel<<<g_elw, 256, 0, stream>>>(y_scan, xi_res, y_bf);

  // 10) out_proj + residual: (4096x2048) x (2048x1024) + x -> out [full tiles]
  {
    dim3 grid(D_MODEL / BN, ROWS / BM);
    wmma_gemm_bf16<true><<<grid, 256, 0, stream>>>(
        y_bf, D_INNER, w_out_bf, D_MODEL, out, D_MODEL,
        ROWS, D_MODEL, D_INNER, 2, nullptr, x);
  }
}
